// SideChainProteinFeatures_67937792688504
// MI455X (gfx1250) — matrix operand order repaired and checked by reading, not compiled, not run on hardware
//
#include <hip/hip_runtime.h>
#include <hip/hip_bf16.h>
#include <math.h>

#define LRES   1024
#define KNN    30
#define NATOMS 14
#define NRBF   16
#define NPOS   16
#define EF     128
#define EDGE_IN 3152          // 16 + 16*14*14
#define KPAD    3168          // padded to multiple of 32
#define NCH     99            // KPAD / 32
#define NROWS   (LRES * KNN)  // 30720

typedef __attribute__((ext_vector_type(16))) _Float16 v16h;
typedef __attribute__((ext_vector_type(8)))  _Float16 v8h;
typedef __attribute__((ext_vector_type(8)))  float    v8f;

union V16 { v16h v; struct { v8h lo, hi; } s; };

// ---------------------------------------------------------------------------
// Kernel 1: masked CA distances + top-30 selection (one block per residue i)
// ---------------------------------------------------------------------------
__global__ __launch_bounds__(256)
void topk_kernel(const float* __restrict__ X, const float* __restrict__ mask,
                 int* __restrict__ Eidx) {
    __shared__ float dist[LRES];
    __shared__ float rval[256];
    __shared__ int   ridx[256];

    const int i   = blockIdx.x;
    const int tid = threadIdx.x;

    const float xi0 = X[(i * NATOMS + 1) * 3 + 0];
    const float xi1 = X[(i * NATOMS + 1) * 3 + 1];
    const float xi2 = X[(i * NATOMS + 1) * 3 + 2];
    const float mi  = mask[i];

    float lmax = 0.0f;
    for (int j = tid; j < LRES; j += 256) {
        float dx = xi0 - X[(j * NATOMS + 1) * 3 + 0];
        float dy = xi1 - X[(j * NATOMS + 1) * 3 + 1];
        float dz = xi2 - X[(j * NATOMS + 1) * 3 + 2];
        float m2 = mi * mask[j];
        float D  = m2 * sqrtf(dx * dx + dy * dy + dz * dz + 1e-6f);
        dist[j]  = D;
        lmax     = fmaxf(lmax, D);
    }
    rval[tid] = lmax;
    __syncthreads();
    for (int s = 128; s > 0; s >>= 1) {
        if (tid < s) rval[tid] = fmaxf(rval[tid], rval[tid + s]);
        __syncthreads();
    }
    const float Dmax = rval[0];
    __syncthreads();

    for (int j = tid; j < LRES; j += 256) {
        float m2 = mi * mask[j];
        dist[j] = dist[j] + 2.0f * (1.0f - m2) * Dmax;
    }
    __syncthreads();

    for (int kk = 0; kk < KNN; ++kk) {
        float bv = INFINITY;
        int   bi = 0x7fffffff;
        for (int j = tid; j < LRES; j += 256) {
            float v = dist[j];
            if (v < bv) { bv = v; bi = j; }
        }
        rval[tid] = bv; ridx[tid] = bi;
        __syncthreads();
        for (int s = 128; s > 0; s >>= 1) {
            if (tid < s) {
                float ov = rval[tid + s]; int oi = ridx[tid + s];
                if (ov < rval[tid] || (ov == rval[tid] && oi < ridx[tid])) {
                    rval[tid] = ov; ridx[tid] = oi;
                }
            }
            __syncthreads();
        }
        if (tid == 0) {
            Eidx[i * KNN + kk] = ridx[0];
            dist[ridx[0]] = INFINITY;
        }
        __syncthreads();
    }
}

// ---------------------------------------------------------------------------
// Kernel 2: build X2 (N,Ca,C,O,Cb,atoms5..13) and am = 1 - atom_mask
// ---------------------------------------------------------------------------
__global__ __launch_bounds__(128)
void build_x2_kernel(const float* __restrict__ X, const float* __restrict__ atom_mask,
                     float* __restrict__ X2w, float* __restrict__ AMw) {
    int i = blockIdx.x * blockDim.x + threadIdx.x;
    if (i >= LRES) return;
    const float* xi = X + i * NATOMS * 3;
    float N0 = xi[0], N1 = xi[1], N2 = xi[2];
    float A0 = xi[3], A1 = xi[4], A2 = xi[5];    // Ca
    float C0 = xi[6], C1 = xi[7], C2 = xi[8];
    float b0 = A0 - N0, b1 = A1 - N1, b2 = A2 - N2;
    float c0 = C0 - A0, c1 = C1 - A1, c2 = C2 - A2;
    float a0 = b1 * c2 - b2 * c1;
    float a1 = b2 * c0 - b0 * c2;
    float a2 = b0 * c1 - b1 * c0;
    float Cb0 = -0.58273431f * a0 + 0.56802827f * b0 - 0.54067466f * c0 + A0;
    float Cb1 = -0.58273431f * a1 + 0.56802827f * b1 - 0.54067466f * c1 + A1;
    float Cb2 = -0.58273431f * a2 + 0.56802827f * b2 - 0.54067466f * c2 + A2;

    float* o = X2w + i * NATOMS * 3;
    // atoms 0..3 unchanged
    for (int a = 0; a < 4; ++a) {
        o[a * 3 + 0] = xi[a * 3 + 0];
        o[a * 3 + 1] = xi[a * 3 + 1];
        o[a * 3 + 2] = xi[a * 3 + 2];
    }
    o[12] = Cb0; o[13] = Cb1; o[14] = Cb2;       // atom 4 = Cb
    for (int a = 5; a < NATOMS; ++a) {
        o[a * 3 + 0] = xi[a * 3 + 0];
        o[a * 3 + 1] = xi[a * 3 + 1];
        o[a * 3 + 2] = xi[a * 3 + 2];
    }
    for (int a = 0; a < NATOMS; ++a)
        AMw[i * NATOMS + a] = 1.0f - atom_mask[i * NATOMS + a];
}

// ---------------------------------------------------------------------------
// Kernel 3: W_edge (128 x 3152, row-major) -> fp16 W16 [KPAD][128], zero pad
// ---------------------------------------------------------------------------
__global__ __launch_bounds__(256)
void wprep_kernel(const float* __restrict__ W, _Float16* __restrict__ W16) {
    int idx = blockIdx.x * 256 + threadIdx.x;
    if (idx >= KPAD * EF) return;
    int k = idx / EF;
    int n = idx % EF;
    float v = (k < EDGE_IN) ? W[n * EDGE_IN + k] : 0.0f;
    W16[idx] = (_Float16)v;
}

// ---------------------------------------------------------------------------
// Kernel 4: fused feature-gen + WMMA GEMM + LayerNorm. 16 edge-rows per block.
// ---------------------------------------------------------------------------
__global__ __launch_bounds__(128)
void edge_kernel(const float* __restrict__ X2w, const float* __restrict__ AMw,
                 const _Float16* __restrict__ W16, const int* __restrict__ Eidx,
                 const float* __restrict__ gamma, const float* __restrict__ beta,
                 float* __restrict__ out) {
    __shared__ _Float16 feat[16][KPAD];         // 101,376 B
    __shared__ float sXi[16][NATOMS][3];
    __shared__ float sXj[16][NATOMS][3];
    __shared__ float sAi[16][NATOMS];
    __shared__ float sAj[16][NATOMS];
    __shared__ float sD[16][EF];                // 8,192 B
    __shared__ int   sI[16], sJ[16];

    const int tid = threadIdx.x;
    const int rb  = blockIdx.x;

    if (tid < 16) {
        int row = rb * 16 + tid;
        sI[tid] = row / KNN;
        sJ[tid] = Eidx[row];
    }
    __syncthreads();

    // stage i-side and j-side atom coordinates + (1 - atom_mask)
    for (int t = tid; t < 2 * 16 * NATOMS; t += 128) {
        int side = t / (16 * NATOMS);
        int rem  = t % (16 * NATOMS);
        int m = rem / NATOMS;
        int a = rem % NATOMS;
        int res = side ? sJ[m] : sI[m];
        float x0 = X2w[(res * NATOMS + a) * 3 + 0];
        float x1 = X2w[(res * NATOMS + a) * 3 + 1];
        float x2 = X2w[(res * NATOMS + a) * 3 + 2];
        float am = AMw[res * NATOMS + a];
        if (side) { sXj[m][a][0] = x0; sXj[m][a][1] = x1; sXj[m][a][2] = x2; sAj[m][a] = am; }
        else      { sXi[m][a][0] = x0; sXi[m][a][1] = x1; sXi[m][a][2] = x2; sAi[m][a] = am; }
    }
    __syncthreads();

    // positional encodings (features 0..15) + zero tail pad (3152..3167)
    for (int t = tid; t < 256; t += 128) {
        int m = t >> 4;
        int f = t & 15;
        float d  = (float)(sJ[m] - sI[m]);
        int   fj = f & 7;
        // freq[j] = exp(-2j * ln(10000)/16)
        float fr = __expf(-(float)(2 * fj) * 0.57564627324851139f);
        float ang = d * fr;
        feat[m][f] = (_Float16)((f < 8) ? __cosf(ang) : __sinf(ang));
        feat[m][EDGE_IN + f] = (_Float16)0.0f;
    }

    // RBF features: 16 rows x 196 atom pairs, 16 gaussians each
    for (int t = tid; t < 16 * NATOMS * NATOMS; t += 128) {
        int m    = t / (NATOMS * NATOMS);
        int pair = t % (NATOMS * NATOMS);
        int a = pair / NATOMS;
        int b = pair % NATOMS;
        float dx = sXi[m][a][0] - sXj[m][b][0];
        float dy = sXi[m][a][1] - sXj[m][b][1];
        float dz = sXi[m][a][2] - sXj[m][b][2];
        float dn = sqrtf(dx * dx + dy * dy + dz * dz + 1e-6f);
        float combo = sAi[m][a] * sAj[m][b];
        _Float16* fp = &feat[m][NPOS + pair * NRBF];
        #pragma unroll
        for (int r = 0; r < NRBF; ++r) {
            float mu = (20.0f / 15.0f) * (float)r;
            float z  = (dn - mu) * 0.8f;          // / sigma, sigma = 1.25
            fp[r] = (_Float16)(__expf(-z * z) * combo);
        }
    }
    __syncthreads();

    // ---- WMMA GEMM: 16 x KPAD (LDS fp16) @ KPAD x 128 (global fp16) ----
    const int lane = tid & 31;
    const int wv   = tid >> 5;       // wave 0..3, handles N-tiles 2w, 2w+1
    const int rowA = lane & 15;
    const int hi   = lane >> 4;      // lane group selector
    const int t0   = wv * 2;
    const int t1   = wv * 2 + 1;

    v8f acc0 = {};
    v8f acc1 = {};
    for (int c = 0; c < NCH; ++c) {
        // A fragment: row = rowA; lanes 0-15 carry K {0..7,16..23}, lanes 16-31 {8..15,24..31}
        int ka = c * 32 + hi * 8;
        V16 A;
        A.s.lo = *(const v8h*)&feat[rowA][ka];
        A.s.hi = *(const v8h*)&feat[rowA][ka + 16];
        // B fragment: N = lane&15 within tile; elements carry same K as A ({0..7,16..23} / {8..15,24..31})
        int kb = c * 32 + hi * 8;
        const _Float16* bp = W16 + kb * EF + rowA;
        V16 B0, B1;
        #pragma unroll
        for (int s = 0; s < 16; ++s) {
            int kk = (s < 8) ? s : s + 8;
            B0.v[s] = bp[kk * EF + t0 * 16];
            B1.v[s] = bp[kk * EF + t1 * 16];
        }
        acc0 = __builtin_amdgcn_wmma_f32_16x16x32_f16(false, A.v, false, B0.v,
                                                      (short)0, acc0, false, false);
        acc1 = __builtin_amdgcn_wmma_f32_16x16x32_f16(false, A.v, false, B1.v,
                                                      (short)0, acc1, false, false);
    }

    // scatter D fragments: VGPR v, lanes 0-15 -> M=v, lanes 16-31 -> M=8+v; N=lane&15
    #pragma unroll
    for (int v = 0; v < 8; ++v) {
        int mrow = hi * 8 + v;
        sD[mrow][t0 * 16 + rowA] = acc0[v];
        sD[mrow][t1 * 16 + rowA] = acc1[v];
    }
    __syncthreads();

    // ---- LayerNorm over 128 features: 8 threads per row, 16 cols each ----
    const int rm  = tid >> 3;
    const int seg = tid & 7;
    float s = 0.0f, sq = 0.0f;
    #pragma unroll
    for (int n = 0; n < 16; ++n) {
        float v = sD[rm][seg * 16 + n];
        s += v; sq += v * v;
    }
    s += __shfl_xor(s, 1);  sq += __shfl_xor(sq, 1);
    s += __shfl_xor(s, 2);  sq += __shfl_xor(sq, 2);
    s += __shfl_xor(s, 4);  sq += __shfl_xor(sq, 4);
    float mean = s * (1.0f / 128.0f);
    float var  = sq * (1.0f / 128.0f) - mean * mean;
    float inv  = rsqrtf(var + 1e-5f);

    int row = rb * 16 + rm;
    float* op = out + row * EF + seg * 16;
    #pragma unroll
    for (int n = 0; n < 16; ++n) {
        int col = seg * 16 + n;
        op[n] = (sD[rm][col] - mean) * inv * gamma[col] + beta[col];
    }
}

// ---------------------------------------------------------------------------
extern "C" void kernel_launch(void* const* d_in, const int* in_sizes, int n_in,
                              void* d_out, int out_size, void* d_ws, size_t ws_size,
                              hipStream_t stream) {
    const float* X         = (const float*)d_in[0];
    const float* mask      = (const float*)d_in[1];
    // d_in[2] residue_idx, d_in[3] chain_labels: unused by the reference math
    const float* atom_mask = (const float*)d_in[4];
    const float* W_edge    = (const float*)d_in[5];
    const float* ln_gamma  = (const float*)d_in[6];
    const float* ln_beta   = (const float*)d_in[7];

    float* out = (float*)d_out;
    // tuple output: E (30720*128 floats) followed by E_idx (30720 int32, bit-cast)
    int* Eidx = (int*)d_out + NROWS * EF;

    // workspace layout
    float*    X2w = (float*)d_ws;                  // 1024*14*3
    float*    AMw = X2w + LRES * NATOMS * 3;       // 1024*14
    _Float16* W16 = (_Float16*)(AMw + LRES * NATOMS); // 3168*128 halfs

    topk_kernel<<<LRES, 256, 0, stream>>>(X, mask, Eidx);
    build_x2_kernel<<<(LRES + 127) / 128, 128, 0, stream>>>(X, atom_mask, X2w, AMw);
    wprep_kernel<<<(KPAD * EF + 255) / 256, 256, 0, stream>>>(W_edge, W16);
    edge_kernel<<<NROWS / 16, 128, 0, stream>>>(X2w, AMw, W16, Eidx,
                                                ln_gamma, ln_beta, out);
}
